// RNN_51479478010496
// MI455X (gfx1250) — matrix operand order repaired and checked by reading, not compiled, not run on hardware
//
#include <hip/hip_runtime.h>
#include <hip/hip_bf16.h>
#include <math.h>

// Problem constants (match reference)
#define B_    1024
#define IN_   256
#define H_    512
#define OUT_  256
#define KTOT  (IN_ + H_)        // 768 combined K for the h-GEMM
#define MTILE 32                // batch rows per workgroup
#define NWG   (B_ / MTILE)      // 32 workgroups
#define ACT_STRIDE 776          // 768 + 8 pad (bank-conflict-free LDS rows)

typedef __bf16 bf16_t;
typedef __attribute__((ext_vector_type(16))) __bf16 v16bf;
typedef __attribute__((ext_vector_type(8)))  __bf16 v8bf;
typedef __attribute__((ext_vector_type(8)))  float  v8f;

// ---------------------------------------------------------------------------
// Fast branch-free tanh.
// Preferred: gfx1250 hardware V_TANH_F32 (trans op) if the builtin exists.
// Fallback: tanh(x) = 1 - 2/(e^{2x}+1) built from raw v_exp_f32 + v_rcp_f32.
// Saturates naturally at +/-1 for large |x| (exp2 -> inf or 0), no EXEC
// divergence (unlike ocml tanhf), ~5 VALU ops per scalar.
// ---------------------------------------------------------------------------
__device__ __forceinline__ float fast_tanh(float x) {
#if __has_builtin(__builtin_amdgcn_tanhf)
    return __builtin_amdgcn_tanhf(x);
#elif __has_builtin(__builtin_amdgcn_tanh_f32)
    return __builtin_amdgcn_tanh_f32(x);
#else
    // 2*log2(e) = 2.885390081777926814
    float e = __builtin_amdgcn_exp2f(x * 2.8853900817779268f);
    return 1.0f - 2.0f * __builtin_amdgcn_rcpf(e + 1.0f);
#endif
}

// ---------------------------------------------------------------------------
// Prologue: convert fp32 weights to bf16, concatenating [W_i2h | W_h2h] into
// Whc (512 x 768, row-major, row = output neuron j) and W_h2o into Woc
// (256 x 512). Row-major by j means a WMMA B-fragment (one column of W^T,
// 32 consecutive K) is a single contiguous 32-byte load per lane.
// ---------------------------------------------------------------------------
__global__ void rnn_prep_weights(const float* __restrict__ Wi2h,
                                 const float* __restrict__ Wh2h,
                                 const float* __restrict__ Wh2o,
                                 bf16_t* __restrict__ Whc,
                                 bf16_t* __restrict__ Woc) {
    int idx = blockIdx.x * blockDim.x + threadIdx.x;
    const int nWh = H_ * KTOT;               // 393216
    if (idx < nWh) {
        int j = idx / KTOT;
        int k = idx - j * KTOT;
        float w = (k < IN_) ? Wi2h[j * IN_ + k] : Wh2h[j * H_ + (k - IN_)];
        Whc[idx] = (bf16_t)w;
    } else {
        int i2 = idx - nWh;
        if (i2 < OUT_ * H_) Woc[i2] = (bf16_t)Wh2o[i2];
    }
}

// ---------------------------------------------------------------------------
// Persistent rollout kernel: each workgroup owns a 32-row batch tile for all
// T steps (output feeds back as input, IN==OUT, so no cross-WG sync needed).
// Activations [x | h] stay in LDS (bf16); weights stream from 192MB L2.
//   h phase: wave w -> N-subtiles 4w..4w+3 of H (j in [64w, 64w+64))
//   y phase: wave w -> N-subtiles 2w..2w+1 of OUT (j in [32w, 32w+32))
// ---------------------------------------------------------------------------
__launch_bounds__(256, 1)
__global__ void rnn_rollout(const float* __restrict__ x0,
                            const float* __restrict__ b_i2h,
                            const float* __restrict__ b_h2h,
                            const float* __restrict__ b_h2o,
                            const bf16_t* __restrict__ Whc,
                            const bf16_t* __restrict__ Woc,
                            float* __restrict__ out,
                            const int* __restrict__ Tp) {
    __shared__ __align__(16) bf16_t s_act[MTILE * ACT_STRIDE];  // [x|h] tile
    __shared__ float s_bsum[H_];     // b_i2h + b_h2h
    __shared__ float s_bo[OUT_];

    const int tid  = threadIdx.x;
    const int lane = tid & 31;
    const int wave = tid >> 5;          // 0..7
    const int half = lane >> 4;         // 0 or 1 (half-wave)
    const int l16  = lane & 15;
    const int b0   = blockIdx.x * MTILE;
    const int T    = *Tp;

    // Biases
    for (int j = tid; j < H_;   j += 256) s_bsum[j] = b_i2h[j] + b_h2h[j];
    for (int j = tid; j < OUT_; j += 256) s_bo[j]   = b_h2o[j];

    // Init activation tile: x region = x_0 (bf16), h region = 0
    for (int i = tid; i < MTILE * KTOT; i += 256) {
        int m = i / KTOT;
        int k = i - m * KTOT;
        float v = (k < IN_) ? x0[(b0 + m) * IN_ + k] : 0.0f;
        s_act[m * ACT_STRIDE + k] = (bf16_t)v;
    }
    // trajectories[0] = x_0
    for (int i = tid; i < MTILE * OUT_; i += 256) {
        int m = i / OUT_;
        int k = i - m * OUT_;
        out[(size_t)(b0 + m) * OUT_ + k] = x0[(b0 + m) * IN_ + k];
    }
    __syncthreads();

    const v8f vzero = {};

    #pragma unroll 1
    for (int t = 1; t < T; ++t) {
        // ================= h phase: [x|h] (32x768) @ Whc^T -> 32x512 ======
        v8f acc[4][2];
        #pragma unroll
        for (int n = 0; n < 4; ++n)
            #pragma unroll
            for (int m = 0; m < 2; ++m) acc[n][m] = vzero;

        #pragma unroll 1
        for (int k0 = 0; k0 < KTOT; k0 += 32) {
            // A fragments (contiguous K order: lane<16 -> K 0..15,
            // lane>=16 -> K 16..31 of its row)
            const int koff = k0 + half * 16;
            v16bf afr[2];
            #pragma unroll
            for (int m = 0; m < 2; ++m) {
                const bf16_t* p = &s_act[(m * 16 + l16) * ACT_STRIDE + koff];
                v8bf lo = *(const v8bf*)p;          // K koff .. koff+7
                v8bf hi = *(const v8bf*)(p + 8);    // K koff+8 .. koff+15
                afr[m] = __builtin_shufflevector(lo, hi,
                          0,1,2,3,4,5,6,7,8,9,10,11,12,13,14,15);
            }
            #pragma unroll
            for (int n = 0; n < 4; ++n) {
                const int j = (wave * 4 + n) * 16 + l16;     // output column
                // B fragment: lane<16 -> K k0..k0+15, lane>=16 -> k0+16..k0+31
                const v16bf bfr = *(const v16bf*)&Whc[(size_t)j * KTOT + k0 + half * 16];
                #pragma unroll
                for (int m = 0; m < 2; ++m)
                    acc[n][m] = __builtin_amdgcn_wmma_f32_16x16x32_bf16(
                        false, afr[m], false, bfr, (short)0, acc[n][m], false, false);
            }
        }

        // bias + tanh, write h (bf16) into act cols [IN_, IN_+H_)
        #pragma unroll
        for (int n = 0; n < 4; ++n) {
            const int j = (wave * 4 + n) * 16 + l16;
            const float bias = s_bsum[j];
            #pragma unroll
            for (int m = 0; m < 2; ++m) {
                #pragma unroll
                for (int r = 0; r < 8; ++r) {
                    // C/D layout: VGPR r -> M = 16m + 8*half + r, N = l16
                    const int row = m * 16 + half * 8 + r;
                    float v = fast_tanh(acc[n][m][r] + bias);
                    s_act[row * ACT_STRIDE + IN_ + j] = (bf16_t)v;
                }
            }
        }
        __syncthreads();

        // ================= y phase: h (32x512) @ Woc^T -> 32x256 ==========
        v8f yacc[2][2];
        #pragma unroll
        for (int n = 0; n < 2; ++n)
            #pragma unroll
            for (int m = 0; m < 2; ++m) yacc[n][m] = vzero;

        #pragma unroll 1
        for (int k0 = 0; k0 < H_; k0 += 32) {
            const int koff = IN_ + k0 + half * 16;
            v16bf afr[2];
            #pragma unroll
            for (int m = 0; m < 2; ++m) {
                const bf16_t* p = &s_act[(m * 16 + l16) * ACT_STRIDE + koff];
                v8bf lo = *(const v8bf*)p;
                v8bf hi = *(const v8bf*)(p + 8);
                afr[m] = __builtin_shufflevector(lo, hi,
                          0,1,2,3,4,5,6,7,8,9,10,11,12,13,14,15);
            }
            #pragma unroll
            for (int n = 0; n < 2; ++n) {
                const int j = (wave * 2 + n) * 16 + l16;
                const v16bf bfr = *(const v16bf*)&Woc[(size_t)j * H_ + k0 + half * 16];
                #pragma unroll
                for (int m = 0; m < 2; ++m)
                    yacc[n][m] = __builtin_amdgcn_wmma_f32_16x16x32_bf16(
                        false, afr[m], false, bfr, (short)0, yacc[n][m], false, false);
            }
        }

        // bias, emit y (fp32 -> HBM) and feed back as next x (bf16 -> LDS)
        const size_t outBase = (size_t)t * B_ * OUT_;
        #pragma unroll
        for (int n = 0; n < 2; ++n) {
            const int j = (wave * 2 + n) * 16 + l16;
            const float bias = s_bo[j];
            #pragma unroll
            for (int m = 0; m < 2; ++m) {
                #pragma unroll
                for (int r = 0; r < 8; ++r) {
                    const int row = m * 16 + half * 8 + r;
                    float v = yacc[n][m][r] + bias;
                    out[outBase + (size_t)(b0 + row) * OUT_ + j] = v;
                    s_act[row * ACT_STRIDE + j] = (bf16_t)v;
                }
            }
        }
        __syncthreads();
    }
}

// ---------------------------------------------------------------------------
extern "C" void kernel_launch(void* const* d_in, const int* in_sizes, int n_in,
                              void* d_out, int out_size, void* d_ws, size_t ws_size,
                              hipStream_t stream) {
    const float* x0   = (const float*)d_in[0];
    const float* Wi2h = (const float*)d_in[1];
    const float* bi2h = (const float*)d_in[2];
    const float* Wh2h = (const float*)d_in[3];
    const float* bh2h = (const float*)d_in[4];
    const float* Wh2o = (const float*)d_in[5];
    const float* bh2o = (const float*)d_in[6];
    const int*   Tp   = (const int*)d_in[7];

    bf16_t* Whc = (bf16_t*)d_ws;                                   // 512x768 bf16
    bf16_t* Woc = (bf16_t*)((char*)d_ws + (size_t)H_ * KTOT * 2);  // 256x512 bf16

    const int totalW = H_ * KTOT + OUT_ * H_;                      // 524288
    rnn_prep_weights<<<totalW / 256, 256, 0, stream>>>(Wi2h, Wh2h, Wh2o, Whc, Woc);
    rnn_rollout<<<NWG, 256, 0, stream>>>(x0, bi2h, bh2h, bh2o, Whc, Woc,
                                         (float*)d_out, Tp);
}
